// Layer_64656437674680
// MI455X (gfx1250) — matrix operand order repaired and checked by reading, not compiled, not run on hardware
//
#include <hip/hip_runtime.h>
#include <hip/hip_bf16.h>
#include <math.h>

#define B_   2
#define S_   2048
#define D_   1024
#define H_   16
#define DH_  64
#define FF_  4096
#define NN_  4096
#define NP_  64
#define K_   16
#define KP_  8
#define BS_  (B_*S_)
#define EPS_ 1e-5f
#define NEG_INF (-3.0e38f)

typedef unsigned short u16;
typedef __attribute__((ext_vector_type(16))) __bf16 v16bf;
typedef __attribute__((ext_vector_type(8)))  float  v8f;

union Frag { v16bf v; uint4 q[2]; };

__device__ __forceinline__ u16 f2bf(float f) {
  unsigned u = __builtin_bit_cast(unsigned, f);
  u += 0x7FFFu + ((u >> 16) & 1u);
  return (u16)(u >> 16);
}
__device__ __forceinline__ float bf2f(u16 h) {
  unsigned u = ((unsigned)h) << 16;
  return __builtin_bit_cast(float, u);
}
__device__ __forceinline__ v8f wmma_bf(const Frag& a, const Frag& b, v8f c) {
  return __builtin_amdgcn_wmma_f32_16x16x32_bf16(false, a.v, false, b.v,
                                                 (short)0, c, false, false);
}

// LDS byte-offset of a __shared__ object (AS3 ptrtoint == offset from LDS base)
typedef __attribute__((address_space(3))) u16 lds_u16;
__device__ __forceinline__ unsigned lds_off(const u16* p) {
  return (unsigned)(unsigned long long)(lds_u16*)p;
}
// Async global->LDS 16B copy per lane (ASYNCcnt-tracked)
__device__ __forceinline__ void async_b128(unsigned ldsoff, unsigned long long gaddr) {
  asm volatile("global_load_async_to_lds_b128 %0, %1, off"
               :: "v"(ldsoff), "v"(gaddr) : "memory");
}

// ---------------------------------------------------------------------------
// fp32 -> bf16 conversion (weights / small tensors)
// ---------------------------------------------------------------------------
extern "C" __global__ void cvt_bf16(const float* __restrict__ src,
                                    u16* __restrict__ dst, int n) {
  int i = blockIdx.x * 256 + threadIdx.x;
  if (i < n) dst[i] = f2bf(src[i]);
}

// ---------------------------------------------------------------------------
// LayerNorm over D=1024, block=256 (4 elems/thread), emit bf16
// ---------------------------------------------------------------------------
extern "C" __global__ void __launch_bounds__(256)
layernorm_bf16(const float* __restrict__ X, const float* __restrict__ G,
               const float* __restrict__ Bt, u16* __restrict__ Out) {
  __shared__ float red[256];
  int t = blockIdx.x, tid = threadIdx.x;
  const float* xr = X + (size_t)t * D_;
  float4 xv = *(const float4*)(xr + tid * 4);
  red[tid] = xv.x + xv.y + xv.z + xv.w;
  __syncthreads();
  for (int st = 128; st; st >>= 1) { if (tid < st) red[tid] += red[tid + st]; __syncthreads(); }
  float mu = red[0] * (1.0f / D_);
  __syncthreads();
  float dx = xv.x - mu, dy = xv.y - mu, dz = xv.z - mu, dw = xv.w - mu;
  red[tid] = dx * dx + dy * dy + dz * dz + dw * dw;
  __syncthreads();
  for (int st = 128; st; st >>= 1) { if (tid < st) red[tid] += red[tid + st]; __syncthreads(); }
  float rstd = rsqrtf(red[0] * (1.0f / D_) + EPS_);
  int d = tid * 4;
  u16* o = Out + (size_t)t * D_ + d;
  o[0] = f2bf(dx * rstd * G[d + 0] + Bt[d + 0]);
  o[1] = f2bf(dy * rstd * G[d + 1] + Bt[d + 1]);
  o[2] = f2bf(dz * rstd * G[d + 2] + Bt[d + 2]);
  o[3] = f2bf(dw * rstd * G[d + 3] + Bt[d + 3]);
}

// ---------------------------------------------------------------------------
// BF16 WMMA GEMM with async-LDS double-buffered staging.
// C[M,N] = A[M,K] * B[N,K]^T.  Block 256 thr = 8 waves (2x4) -> 64x256 tile,
// K-step 64, LDS: [2 bufs][A 64x72 | B 256x72] bf16 (padded rows, 16B chunks).
// mode bits: 2=+residual, 4=f32 out, 8=bf16 out row-major,
//            16=bf16 out V-transposed, 32=+bias
// ---------------------------------------------------------------------------
#define APAD_ 72
#define ABUF_ (64 * APAD_)            // 4608 shorts
#define BBUF_ (256 * APAD_)           // 18432 shorts
#define BUFSH_ (ABUF_ + BBUF_)        // 23040 shorts per buffer

extern "C" __global__ void __launch_bounds__(256)
gemm_async(const u16* __restrict__ A, const u16* __restrict__ Bm,
           const float* __restrict__ bias, const float* __restrict__ residual,
           float* __restrict__ outF, u16* __restrict__ outBf,
           u16* __restrict__ outVt, int M, int N, int Kd, int mode) {
  __shared__ u16 smem[2 * BUFSH_];
  const int tid = threadIdx.x;
  const int lane = tid & 31, wave = tid >> 5;
  const int half = lane >> 4, l16 = lane & 15;
  const int row0 = blockIdx.y * 64;
  const int col0 = blockIdx.x * 256;
  const int wr = wave >> 2, wc = wave & 3;
  const unsigned smemBase = lds_off(smem);

  // staging maps: 16B chunks; A: 64 rows x 8 chunks (2/thread),
  // B: 256 rows x 8 chunks (8/thread); consecutive tids -> contiguous 128B runs
  unsigned long long gaA[2]; unsigned ldsA[2];
#pragma unroll
  for (int j = 0; j < 2; ++j) {
    int chunk = tid + j * 256, r = chunk >> 3, c = chunk & 7;
    gaA[j] = (unsigned long long)(A + (size_t)(row0 + r) * Kd + c * 8);
    ldsA[j] = (unsigned)((r * APAD_ + c * 8) * 2);
  }
  unsigned long long gaB[8]; unsigned ldsB[8];
#pragma unroll
  for (int j = 0; j < 8; ++j) {
    int chunk = tid + j * 256, r = chunk >> 3, c = chunk & 7;
    gaB[j] = (unsigned long long)(Bm + (size_t)(col0 + r) * Kd + c * 8);
    ldsB[j] = (unsigned)((ABUF_ + r * APAD_ + c * 8) * 2);
  }

  v8f acc[2][4] = {};

  // prologue: stage buffer 0
#pragma unroll
  for (int j = 0; j < 2; ++j) async_b128(smemBase + ldsA[j], gaA[j]);
#pragma unroll
  for (int j = 0; j < 8; ++j) async_b128(smemBase + ldsB[j], gaB[j]);

  int buf = 0;
  for (int kt = 0; kt < Kd; kt += 64) {
    if (kt + 64 < Kd) {
      const unsigned bo = smemBase + (unsigned)((buf ^ 1) * BUFSH_ * 2);
      const unsigned long long gk = (unsigned long long)(kt + 64) * 2;
#pragma unroll
      for (int j = 0; j < 2; ++j) async_b128(bo + ldsA[j], gaA[j] + gk);
#pragma unroll
      for (int j = 0; j < 8; ++j) async_b128(bo + ldsB[j], gaB[j] + gk);
      asm volatile("s_wait_asynccnt 0xa" ::: "memory");  // current buffer done
    } else {
      asm volatile("s_wait_asynccnt 0x0" ::: "memory");
    }
    __syncthreads();
    const u16* sa = smem + buf * BUFSH_ + (wr * 32 + l16) * APAD_;
    const u16* sb = smem + buf * BUFSH_ + ABUF_ + (wc * 64 + l16) * APAD_;
#pragma unroll
    for (int k0 = 0; k0 < 64; k0 += 32) {
      Frag a0, a1;
      const u16* p = sa + k0 + half * 8;
      a0.q[0] = *(const uint4*)p; a0.q[1] = *(const uint4*)(p + 16);
      p = sa + 16 * APAD_ + k0 + half * 8;
      a1.q[0] = *(const uint4*)p; a1.q[1] = *(const uint4*)(p + 16);
      Frag bfr[4];
#pragma unroll
      for (int ci = 0; ci < 4; ++ci) {
        p = sb + ci * 16 * APAD_ + k0 + half * 8;
        bfr[ci].q[0] = *(const uint4*)p; bfr[ci].q[1] = *(const uint4*)(p + 16);
      }
#pragma unroll
      for (int ci = 0; ci < 4; ++ci) {
        acc[0][ci] = wmma_bf(a0, bfr[ci], acc[0][ci]);
        acc[1][ci] = wmma_bf(a1, bfr[ci], acc[1][ci]);
      }
    }
    __syncthreads();
    buf ^= 1;
  }

  const int rbase = row0 + wr * 32, cbase = col0 + wc * 64;
#pragma unroll
  for (int ri = 0; ri < 2; ++ri) {
#pragma unroll
    for (int v = 0; v < 8; ++v) {
      const int m = rbase + ri * 16 + v + 8 * half;
#pragma unroll
      for (int ci = 0; ci < 4; ++ci) {
        const int n = cbase + ci * 16 + l16;
        float val = acc[ri][ci][v];
        if (mode & 32) val += bias[n];
        if (mode & 2)  val += residual[(size_t)m * N + n];
        if (mode & 4)  outF[(size_t)m * N + n] = val;
        if (mode & 8)  outBf[(size_t)m * N + n] = f2bf(val);
        if (mode & 16) {
          int bb = m >> 11, s = m & (S_ - 1), hh = n >> 6, d = n & (DH_ - 1);
          outVt[(((size_t)bb * H_ + hh) * DH_ + d) * S_ + s] = f2bf(val);
        }
      }
    }
  }
}

// ---------------------------------------------------------------------------
// Direct-global BF16 WMMA GEMM used for the dual-A fused router scores:
// C = pw0[m]*(A@B^T) + pw1[m]*(A2@B^T).  mode bits as gemm_async, +1=dual-A.
// ---------------------------------------------------------------------------
extern "C" __global__ void __launch_bounds__(256)
gemm_bf16(const u16* __restrict__ A, const u16* __restrict__ A2,
          const u16* __restrict__ Bm, const float* __restrict__ bias,
          const float* __restrict__ pw, const float* __restrict__ residual,
          float* __restrict__ outF, u16* __restrict__ outBf,
          u16* __restrict__ outVt, int M, int N, int Kd, int mode) {
  const int lane = threadIdx.x & 31;
  const int wave = threadIdx.x >> 5;
  const int half = lane >> 4, l16 = lane & 15;
  const int row0 = blockIdx.y * 64 + (wave >> 2) * 32;
  const int col0 = blockIdx.x * 256 + (wave & 3) * 64;
  const size_t aOff0 = (size_t)(row0 + l16) * Kd + half * 8;
  const size_t aOff1 = (size_t)(row0 + 16 + l16) * Kd + half * 8;
  size_t bOff[4];
#pragma unroll
  for (int ci = 0; ci < 4; ++ci)
    bOff[ci] = (size_t)(col0 + ci * 16 + l16) * Kd + half * 8;

  v8f acc[2][4] = {}; v8f acc2[2][4] = {};
  const bool dual = (mode & 1);

  for (int k0 = 0; k0 < Kd; k0 += 32) {
    Frag a0, a1, bfr[4];
    const u16* p = A + aOff0 + k0;
    a0.q[0] = *(const uint4*)p; a0.q[1] = *(const uint4*)(p + 16);
    p = A + aOff1 + k0;
    a1.q[0] = *(const uint4*)p; a1.q[1] = *(const uint4*)(p + 16);
#pragma unroll
    for (int ci = 0; ci < 4; ++ci) {
      p = Bm + bOff[ci] + k0;
      bfr[ci].q[0] = *(const uint4*)p; bfr[ci].q[1] = *(const uint4*)(p + 16);
    }
#pragma unroll
    for (int ci = 0; ci < 4; ++ci) {
      acc[0][ci] = wmma_bf(a0, bfr[ci], acc[0][ci]);
      acc[1][ci] = wmma_bf(a1, bfr[ci], acc[1][ci]);
    }
    if (dual) {
      Frag c0, c1;
      p = A2 + aOff0 + k0;
      c0.q[0] = *(const uint4*)p; c0.q[1] = *(const uint4*)(p + 16);
      p = A2 + aOff1 + k0;
      c1.q[0] = *(const uint4*)p; c1.q[1] = *(const uint4*)(p + 16);
#pragma unroll
      for (int ci = 0; ci < 4; ++ci) {
        acc2[0][ci] = wmma_bf(c0, bfr[ci], acc2[0][ci]);
        acc2[1][ci] = wmma_bf(c1, bfr[ci], acc2[1][ci]);
      }
    }
  }

#pragma unroll
  for (int ri = 0; ri < 2; ++ri) {
#pragma unroll
    for (int v = 0; v < 8; ++v) {
      const int m = row0 + ri * 16 + v + 8 * half;
      float s0 = 0.f, s1 = 0.f;
      if (dual) { s0 = pw[2 * m]; s1 = pw[2 * m + 1]; }
#pragma unroll
      for (int ci = 0; ci < 4; ++ci) {
        const int n = col0 + ci * 16 + l16;
        float val = acc[ri][ci][v];
        if (dual) val = s0 * val + s1 * acc2[ri][ci][v];
        if (mode & 32) val += bias[n];
        if (mode & 2)  val += residual[(size_t)m * N + n];
        if (mode & 4)  outF[(size_t)m * N + n] = val;
        if (mode & 8)  outBf[(size_t)m * N + n] = f2bf(val);
        if (mode & 16) {
          int bb = m >> 11, s = m & (S_ - 1), hh = n >> 6, d = n & (DH_ - 1);
          outVt[(((size_t)bb * H_ + hh) * DH_ + d) * S_ + s] = f2bf(val);
        }
      }
    }
  }
}

// ---------------------------------------------------------------------------
// Flash attention: block 128 = 4 waves, each wave owns a 16-query tile.
// QK^T and P*V via bf16 WMMA; online softmax with intra-half shuffles;
// P relaid C->A through per-wave LDS guarded by s_wait_dscnt.
// ---------------------------------------------------------------------------
extern "C" __global__ void __launch_bounds__(128)
attn_flash(const u16* __restrict__ Q, const u16* __restrict__ Kb,
           const u16* __restrict__ Vt, u16* __restrict__ Ctx) {
  __shared__ u16 pls[4][16 * 32];
  const int lane = threadIdx.x & 31;
  const int wave = threadIdx.x >> 5;
  const int half = lane >> 4, l16 = lane & 15;
  const int bh = blockIdx.y;
  const int b = bh / H_, h = bh % H_;
  const int qt = blockIdx.x * 4 + wave;

  const u16* qrow = Q + (size_t)(b * S_ + qt * 16 + l16) * D_ + h * DH_ + half * 16;
  Frag aq0, aq1;
  aq0.q[0] = *(const uint4*)qrow;        aq0.q[1] = *(const uint4*)(qrow + 8);
  aq1.q[0] = *(const uint4*)(qrow + 32); aq1.q[1] = *(const uint4*)(qrow + 40);

  v8f o0 = {}, o1 = {}, o2 = {}, o3 = {};
  float mrow[8], lrow[8];
#pragma unroll
  for (int v = 0; v < 8; ++v) { mrow[v] = NEG_INF; lrow[v] = 0.f; }

  const size_t kBase = (size_t)(b * S_) * D_ + h * DH_;
  const size_t vBase = (size_t)(b * H_ + h) * DH_ * S_;
  u16* pb = &pls[wave][0];

  for (int kb = 0; kb < S_; kb += 32) {
    v8f s0 = {}, s1 = {};
    {
      const u16* kr = Kb + kBase + (size_t)(kb + l16) * D_ + half * 16;
      Frag b0, b1;
      b0.q[0] = *(const uint4*)kr;        b0.q[1] = *(const uint4*)(kr + 8);
      b1.q[0] = *(const uint4*)(kr + 32); b1.q[1] = *(const uint4*)(kr + 40);
      s0 = wmma_bf(aq0, b0, s0);
      s0 = wmma_bf(aq1, b1, s0);
    }
    {
      const u16* kr = Kb + kBase + (size_t)(kb + 16 + l16) * D_ + half * 16;
      Frag b0, b1;
      b0.q[0] = *(const uint4*)kr;        b0.q[1] = *(const uint4*)(kr + 8);
      b1.q[0] = *(const uint4*)(kr + 32); b1.q[1] = *(const uint4*)(kr + 40);
      s1 = wmma_bf(aq0, b0, s1);
      s1 = wmma_bf(aq1, b1, s1);
    }
    float al[8];
#pragma unroll
    for (int v = 0; v < 8; ++v) {
      float x0 = s0[v] * 0.125f, x1 = s1[v] * 0.125f;   // 1/sqrt(DH)
      float mx = fmaxf(x0, x1);
      mx = fmaxf(mx, __shfl_xor(mx, 1, 32));
      mx = fmaxf(mx, __shfl_xor(mx, 2, 32));
      mx = fmaxf(mx, __shfl_xor(mx, 4, 32));
      mx = fmaxf(mx, __shfl_xor(mx, 8, 32));
      float mn = fmaxf(mrow[v], mx);
      float a  = __expf(mrow[v] - mn);
      float e0 = __expf(x0 - mn), e1 = __expf(x1 - mn);
      float rs = e0 + e1;
      rs += __shfl_xor(rs, 1, 32);
      rs += __shfl_xor(rs, 2, 32);
      rs += __shfl_xor(rs, 4, 32);
      rs += __shfl_xor(rs, 8, 32);
      mrow[v] = mn;
      lrow[v] = lrow[v] * a + rs;
      al[v] = a;
      s0[v] = e0; s1[v] = e1;
    }
#pragma unroll
    for (int v = 0; v < 8; ++v) {
      o0[v] *= al[v]; o1[v] *= al[v]; o2[v] *= al[v]; o3[v] *= al[v];
    }
    // P (C-layout) -> LDS -> A-fragment layout
#pragma unroll
    for (int v = 0; v < 8; ++v) {
      pb[(v + 8 * half) * 32 + l16]      = f2bf(s0[v]);
      pb[(v + 8 * half) * 32 + 16 + l16] = f2bf(s1[v]);
    }
    asm volatile("s_wait_dscnt 0" ::: "memory");
    Frag pa;
    const u16* pr = pb + l16 * 32 + half * 16;
    pa.q[0] = *(const uint4*)pr; pa.q[1] = *(const uint4*)(pr + 8);
#pragma unroll
    for (int di = 0; di < 4; ++di) {
      const u16* vr = Vt + vBase + (size_t)(di * 16 + l16) * S_ + kb + half * 16;
      Frag bv;
      bv.q[0] = *(const uint4*)vr; bv.q[1] = *(const uint4*)(vr + 8);
      if (di == 0) o0 = wmma_bf(pa, bv, o0);
      if (di == 1) o1 = wmma_bf(pa, bv, o1);
      if (di == 2) o2 = wmma_bf(pa, bv, o2);
      if (di == 3) o3 = wmma_bf(pa, bv, o3);
    }
  }
#pragma unroll
  for (int v = 0; v < 8; ++v) {
    float inv = 1.0f / lrow[v];
    int sq = qt * 16 + v + 8 * half;
    u16* o = Ctx + (size_t)(b * S_ + sq) * D_ + h * DH_;
    o[l16]      = f2bf(o0[v] * inv);
    o[16 + l16] = f2bf(o1[v] * inv);
    o[32 + l16] = f2bf(o2[v] * inv);
    o[48 + l16] = f2bf(o3[v] * inv);
  }
}

// ---------------------------------------------------------------------------
// Path mixing weights: softmax( [normed,ctx] @ wp^T + bp ) per token
// ---------------------------------------------------------------------------
extern "C" __global__ void __launch_bounds__(256)
path_weights(const u16* __restrict__ Nrm, const u16* __restrict__ Ctx,
             const float* __restrict__ Wp, const float* __restrict__ Bp,
             float* __restrict__ Pw) {
  __shared__ float r0[256], r1[256];
  int t = blockIdx.x, tid = threadIdx.x;
  const u16* nr = Nrm + (size_t)t * D_;
  const u16* cr = Ctx + (size_t)t * D_;
  float a0 = 0.f, a1 = 0.f;
#pragma unroll
  for (int j = 0; j < 4; ++j) {
    int d = tid * 4 + j;
    float nv = bf2f(nr[d]), cv = bf2f(cr[d]);
    a0 += nv * Wp[d] + cv * Wp[D_ + d];
    a1 += nv * Wp[2 * D_ + d] + cv * Wp[3 * D_ + d];
  }
  r0[tid] = a0; r1[tid] = a1; __syncthreads();
  for (int st = 128; st; st >>= 1) {
    if (tid < st) { r0[tid] += r0[tid + st]; r1[tid] += r1[tid + st]; }
    __syncthreads();
  }
  if (tid == 0) {
    float l0 = r0[0] + Bp[0], l1 = r1[0] + Bp[1];
    float mx = fmaxf(l0, l1);
    float e0 = __expf(l0 - mx), e1 = __expf(l1 - mx);
    float inv = 1.0f / (e0 + e1);
    Pw[2 * t] = e0 * inv; Pw[2 * t + 1] = e1 * inv;
  }
}

// ---------------------------------------------------------------------------
// Top-16 over 4096 neuron scores (one block per token) + softmax weights
// ---------------------------------------------------------------------------
extern "C" __global__ void __launch_bounds__(256)
topk_router(const float* __restrict__ Sc, float* __restrict__ Tw,
            int* __restrict__ TiW, int* __restrict__ TiOut) {
  __shared__ float sv[NN_];
  __shared__ float rv[256];
  __shared__ int   ri[256];
  __shared__ float kv[K_];
  __shared__ int   ki[K_];
  int t = blockIdx.x, tid = threadIdx.x;
  const float* sr = Sc + (size_t)t * NN_;
  for (int i = tid; i < NN_; i += 256) sv[i] = sr[i];
  __syncthreads();
  for (int r = 0; r < K_; ++r) {
    float bv = NEG_INF; int bi = 0x7fffffff;
    for (int i = tid; i < NN_; i += 256) {
      float v = sv[i];
      if (v > bv) { bv = v; bi = i; }
    }
    rv[tid] = bv; ri[tid] = bi; __syncthreads();
    for (int st = 128; st; st >>= 1) {
      if (tid < st) {
        float ov = rv[tid + st]; int oi = ri[tid + st];
        if (ov > rv[tid] || (ov == rv[tid] && oi < ri[tid])) { rv[tid] = ov; ri[tid] = oi; }
      }
      __syncthreads();
    }
    if (tid == 0) { kv[r] = rv[0]; ki[r] = ri[0]; sv[ri[0]] = NEG_INF; }
    __syncthreads();
  }
  if (tid == 0) {
    float mx = kv[0], s = 0.f, e[K_];
#pragma unroll
    for (int k = 0; k < K_; ++k) { e[k] = __expf(kv[k] - mx); s += e[k]; }
    float inv = 1.0f / s;
#pragma unroll
    for (int k = 0; k < K_; ++k) {
      Tw[t * K_ + k] = e[k] * inv;
      TiW[t * K_ + k] = ki[k];
      TiOut[t * K_ + k] = ki[k];
    }
  }
}

// ---------------------------------------------------------------------------
// router_out = sum_k w_k * neurons[idx_k];  xnew = x + router_out
// ---------------------------------------------------------------------------
extern "C" __global__ void __launch_bounds__(256)
router_gather(const float* __restrict__ X, const float* __restrict__ Neur,
              const float* __restrict__ Tw, const int* __restrict__ Ti,
              float* __restrict__ Xn, u16* __restrict__ Rb) {
  __shared__ float w[K_];
  __shared__ int  id[K_];
  int t = blockIdx.x, tid = threadIdx.x;
  if (tid < K_) { w[tid] = Tw[t * K_ + tid]; id[tid] = Ti[t * K_ + tid]; }
  __syncthreads();
  for (int d = tid; d < D_; d += 256) {
    float acc = 0.f;
#pragma unroll
    for (int k = 0; k < K_; ++k) acc += w[k] * Neur[(size_t)id[k] * D_ + d];
    Xn[(size_t)t * D_ + d] = X[(size_t)t * D_ + d] + acc;
    Rb[(size_t)t * D_ + d] = f2bf(acc);
  }
}

// ---------------------------------------------------------------------------
// pattern_scores[t,p] = dot(router_out[t], pattern_queries[p]) / sqrt(D)
// ---------------------------------------------------------------------------
extern "C" __global__ void __launch_bounds__(64)
pattern_scores(const u16* __restrict__ Rb, const u16* __restrict__ Pq,
               float* __restrict__ Ps) {
  int t = blockIdx.x, p = threadIdx.x;
  const u16* rr = Rb + (size_t)t * D_;
  const u16* qr = Pq + (size_t)p * D_;
  float acc = 0.f;
  for (int d = 0; d < D_; ++d) acc += bf2f(rr[d]) * bf2f(qr[d]);
  Ps[(size_t)t * NP_ + p] = acc * 0.03125f;   // 1/sqrt(1024)
}

// ---------------------------------------------------------------------------
// Top-8 of 64 pattern scores per token (one wave/token) + softmax
// ---------------------------------------------------------------------------
extern "C" __global__ void __launch_bounds__(256)
pattern_topk(const float* __restrict__ Ps, float* __restrict__ Pw,
             int* __restrict__ Pi) {
  int t = blockIdx.x * 8 + (threadIdx.x >> 5);
  int lane = threadIdx.x & 31;
  const float* r = Ps + (size_t)t * NP_;
  float v0 = r[lane], v1 = r[lane + 32];
  float sv[KP_]; int si[KP_];
#pragma unroll
  for (int k = 0; k < KP_; ++k) {
    float bv; int bi;
    if (v0 >= v1) { bv = v0; bi = lane; } else { bv = v1; bi = lane + 32; }
    for (int off = 16; off; off >>= 1) {
      float ov = __shfl_xor(bv, off, 32);
      int   oi = __shfl_xor(bi, off, 32);
      if (ov > bv || (ov == bv && oi < bi)) { bv = ov; bi = oi; }
    }
    sv[k] = bv; si[k] = bi;
    if (bi == lane)      v0 = NEG_INF;
    if (bi == lane + 32) v1 = NEG_INF;
  }
  float mx = sv[0], s = 0.f, e[KP_];
#pragma unroll
  for (int k = 0; k < KP_; ++k) { e[k] = __expf(sv[k] - mx); s += e[k]; }
  if (lane < KP_) {
    Pw[(size_t)t * KP_ + lane] = e[lane] / s;
    Pi[(size_t)t * KP_ + lane] = si[lane];
  }
}

// ---------------------------------------------------------------------------
// ffn_gate gather + h*sigmoid(gate) + exact GELU -> bf16
// ---------------------------------------------------------------------------
extern "C" __global__ void __launch_bounds__(256)
gate_gelu(const float* __restrict__ Hb, const float* __restrict__ Gates,
          const float* __restrict__ Pw, const int* __restrict__ Pi,
          u16* __restrict__ Hg) {
  __shared__ float w[KP_];
  __shared__ int  id[KP_];
  int t = blockIdx.x, tid = threadIdx.x;
  if (tid < KP_) { w[tid] = Pw[t * KP_ + tid]; id[tid] = Pi[t * KP_ + tid]; }
  __syncthreads();
  for (int f = tid; f < FF_; f += 256) {
    float g = 0.f;
#pragma unroll
    for (int k = 0; k < KP_; ++k) g += w[k] * Gates[(size_t)id[k] * FF_ + f];
    float hv = Hb[(size_t)t * FF_ + f];
    float z  = hv * (1.0f / (1.0f + __expf(-g)));
    float ge = 0.5f * z * (1.0f + erff(z * 0.70710678118654752f));
    Hg[(size_t)t * FF_ + f] = f2bf(ge);
  }
}

// ---------------------------------------------------------------------------
// Host orchestration
// ---------------------------------------------------------------------------
extern "C" void kernel_launch(void* const* d_in, const int* in_sizes, int n_in,
                              void* d_out, int out_size, void* d_ws, size_t ws_size,
                              hipStream_t stream) {
  (void)in_sizes; (void)n_in; (void)out_size; (void)ws_size;
  const float* x       = (const float*)d_in[0];
  const float* neurons = (const float*)d_in[1];
  const float* wq      = (const float*)d_in[2];
  const float* bq      = (const float*)d_in[3];
  const float* wk      = (const float*)d_in[4];
  const float* bk      = (const float*)d_in[5];
  const float* wv      = (const float*)d_in[6];
  const float* bv      = (const float*)d_in[7];
  const float* wp      = (const float*)d_in[8];
  const float* bp      = (const float*)d_in[9];
  const float* pq      = (const float*)d_in[10];
  const float* gates   = (const float*)d_in[11];
  const float* w_up    = (const float*)d_in[12];
  const float* b_up    = (const float*)d_in[13];
  const float* w_down  = (const float*)d_in[14];
  const float* b_down  = (const float*)d_in[15];
  const float* ln1g    = (const float*)d_in[16];
  const float* ln1b    = (const float*)d_in[17];
  const float* ln2g    = (const float*)d_in[18];
  const float* ln2b    = (const float*)d_in[19];

  char* ws = (char*)d_ws;
  size_t off = 0;
  auto alloc = [&](size_t bytes) -> char* {
    char* r = ws + off;
    off = (off + bytes + 255) & ~(size_t)255;
    return r;
  };
  u16* neur_bf    = (u16*)alloc((size_t)NN_ * D_ * 2);
  u16* wq_bf      = (u16*)alloc((size_t)D_ * D_ * 2);
  u16* wk_bf      = (u16*)alloc((size_t)D_ * D_ * 2);
  u16* wv_bf      = (u16*)alloc((size_t)D_ * D_ * 2);
  u16* pq_bf      = (u16*)alloc((size_t)NP_ * D_ * 2);
  u16* wup_bf     = (u16*)alloc((size_t)FF_ * D_ * 2);
  u16* wdn_bf     = (u16*)alloc((size_t)D_ * FF_ * 2);
  u16* normed_bf  = (u16*)alloc((size_t)BS_ * D_ * 2);
  u16* q_bf       = (u16*)alloc((size_t)BS_ * D_ * 2);
  u16* k_bf       = (u16*)alloc((size_t)BS_ * D_ * 2);
  u16* vt_bf      = (u16*)alloc((size_t)BS_ * D_ * 2);
  u16* ctx_bf     = (u16*)alloc((size_t)BS_ * D_ * 2);
  float* pwbuf    = (float*)alloc((size_t)BS_ * 2 * 4);
  float* xnew     = (float*)alloc((size_t)BS_ * D_ * 4);
  u16* rout_bf    = (u16*)alloc((size_t)BS_ * D_ * 2);
  u16* normed2_bf = (u16*)alloc((size_t)BS_ * D_ * 2);
  float* psbuf    = (float*)alloc((size_t)BS_ * NP_ * 4);
  float* patw     = (float*)alloc((size_t)BS_ * KP_ * 4);
  int*   pati     = (int*)alloc((size_t)BS_ * KP_ * 4);
  float* topw     = (float*)alloc((size_t)BS_ * K_ * 4);
  int*   topidx   = (int*)alloc((size_t)BS_ * K_ * 4);
  float* bigbuf   = (float*)alloc((size_t)BS_ * NN_ * 4);  // scores, then h (reused)
  u16* hg_bf      = (u16*)alloc((size_t)BS_ * FF_ * 2);

  float* outF   = (float*)d_out;
  int*   outIdx = (int*)(outF + (size_t)BS_ * D_);

  // weight conversions (fp32 -> bf16)
  auto cvt = [&](const float* s, u16* d, int n) {
    cvt_bf16<<<(n + 255) / 256, 256, 0, stream>>>(s, d, n);
  };
  cvt(neurons, neur_bf, NN_ * D_);
  cvt(wq, wq_bf, D_ * D_);
  cvt(wk, wk_bf, D_ * D_);
  cvt(wv, wv_bf, D_ * D_);
  cvt(pq, pq_bf, NP_ * D_);
  cvt(w_up, wup_bf, FF_ * D_);
  cvt(w_down, wdn_bf, D_ * FF_);

  // LN1
  layernorm_bf16<<<BS_, 256, 0, stream>>>(x, ln1g, ln1b, normed_bf);

  // QKV projections (async-LDS staged WMMA GEMMs)
  dim3 gD(D_ / 256, BS_ / 64);
  gemm_async<<<gD, 256, 0, stream>>>(normed_bf, wq_bf, bq, nullptr,
                                     nullptr, q_bf, nullptr, BS_, D_, D_, 8 | 32);
  gemm_async<<<gD, 256, 0, stream>>>(normed_bf, wk_bf, bk, nullptr,
                                     nullptr, k_bf, nullptr, BS_, D_, D_, 8 | 32);
  gemm_async<<<gD, 256, 0, stream>>>(normed_bf, wv_bf, bv, nullptr,
                                     nullptr, nullptr, vt_bf, BS_, D_, D_, 16 | 32);

  // attention
  attn_flash<<<dim3(S_ / 64, B_ * H_), 128, 0, stream>>>(q_bf, k_bf, vt_bf, ctx_bf);

  // path mixing weights
  path_weights<<<BS_, 256, 0, stream>>>(normed_bf, ctx_bf, wp, bp, pwbuf);

  // fused dual-A router scores: pw0*normed@N^T + pw1*ctx@N^T
  dim3 gS(NN_ / 256, BS_ / 64);
  gemm_bf16<<<gS, 256, 0, stream>>>(normed_bf, ctx_bf, neur_bf, nullptr, pwbuf, nullptr,
                                    bigbuf, nullptr, nullptr, BS_, NN_, D_, 1 | 4);

  // top-16 + softmax; router gather + residual
  topk_router<<<BS_, 256, 0, stream>>>(bigbuf, topw, topidx, outIdx);
  router_gather<<<BS_, 256, 0, stream>>>(x, neurons, topw, topidx, xnew, rout_bf);

  // LN2
  layernorm_bf16<<<BS_, 256, 0, stream>>>(xnew, ln2g, ln2b, normed2_bf);

  // pattern routing
  pattern_scores<<<BS_, 64, 0, stream>>>(rout_bf, pq_bf, psbuf);
  pattern_topk<<<BS_ / 8, 256, 0, stream>>>(psbuf, patw, pati);

  // FFN up (h into bigbuf, reusing scores space)
  dim3 gU(FF_ / 256, BS_ / 64);
  gemm_async<<<gU, 256, 0, stream>>>(normed2_bf, wup_bf, b_up, nullptr,
                                     bigbuf, nullptr, nullptr, BS_, FF_, D_, 4 | 32);

  // gate + gelu
  gate_gelu<<<BS_, 256, 0, stream>>>(bigbuf, gates, patw, pati, hg_bf);

  // FFN down + residual -> d_out
  gemm_async<<<gD, 256, 0, stream>>>(hg_bf, wdn_bf, b_down, xnew,
                                     outF, nullptr, nullptr, BS_, D_, FF_, 2 | 4 | 32);
}
